// StatePredictorModel_1142461301083
// MI455X (gfx1250) — matrix operand order repaired and checked by reading, not compiled, not run on hardware
//
#include <hip/hip_runtime.h>

typedef _Float16 half_t;
typedef _Float16 v16h  __attribute__((ext_vector_type(16)));
typedef float    v8f   __attribute__((ext_vector_type(8)));
typedef unsigned u32x4 __attribute__((ext_vector_type(4)));
typedef int      i32x4 __attribute__((ext_vector_type(4)));

typedef __attribute__((address_space(1))) i32x4* as1_i4p;  // global int4*
typedef __attribute__((address_space(3))) i32x4* as3_i4p;  // LDS int4*

#if defined(__has_builtin)
#if __has_builtin(__builtin_amdgcn_global_load_async_to_lds_b128)
#define HAVE_ASYNC_LDS 1
#endif
#endif

// ---------------------------------------------------------------------------
// WMMA tile loaders (CDNA5 16-bit layouts, ISA 7.12.2)
// A 16x32 (MxK): lane L row = (L&15); elems 0..7 <-> K=kb..kb+7,
//   elems 8..15 <-> K=kb+16..kb+23, kb = kt*32 + ((L>>4)&1)*8.
// B 32x16 (KxN) = W^T of row-major W[n][k]: lane L col n = (L&15);
//   same K order as A: elems 0..7 <-> K=kb..kb+7, elems 8..15 <-> K=kb+16..kb+23.
// ---------------------------------------------------------------------------
__device__ __forceinline__ v16h load_a_tile(const half_t* __restrict__ A, int lda,
                                            int row, int kt, int lane) {
  const half_t* p = A + (size_t)row * lda + (size_t)kt * 32 + ((lane >> 4) & 1) * 8;
  union { v16h v; u32x4 q[2]; } u;
  u.q[0] = *(const u32x4*)(p);
  u.q[1] = *(const u32x4*)(p + 16);
  return u.v;
}

__device__ __forceinline__ v16h load_b_tile(const half_t* __restrict__ W, int ldw,
                                            int nrow, int kt, int lane) {
  const half_t* p = W + (size_t)nrow * ldw + (size_t)kt * 32 + ((lane >> 4) & 1) * 8;
  union { v16h v; u32x4 q[2]; } u;
  u.q[0] = *(const u32x4*)(p);
  u.q[1] = *(const u32x4*)(p + 16);
  return u.v;
}

__device__ __forceinline__ float sigf(float x) { return 1.0f / (1.0f + __expf(-x)); }

// ---------------------------------------------------------------------------
// Generic WMMA GEMM: C[m][n] = A[m][:K] . W[n][:K] + bias0[n] (+ bias1[n])
// wave tile = 16(M) x 64(N); 8 waves/block, exact coverage.
// Cf32 != nullptr -> f32 + bias0+bias1 ; else relu + f16 + bias0.
// ---------------------------------------------------------------------------
__global__ void __launch_bounds__(256) wmma_gemm_bias(
    const half_t* __restrict__ A, const half_t* __restrict__ W,
    const float* __restrict__ bias0, const float* __restrict__ bias1,
    float* __restrict__ Cf32, half_t* __restrict__ Cf16,
    int N, int K, int n64tiles) {
  const int lane = threadIdx.x & 31;
  const int wid  = blockIdx.x * 8 + (threadIdx.x >> 5);
  const int m16  = wid / n64tiles;
  const int n64  = wid % n64tiles;
  const int arow = m16 * 16 + (lane & 15);
  const int kTiles = K >> 5;

  v8f acc[4];
  acc[0] = v8f{}; acc[1] = v8f{}; acc[2] = v8f{}; acc[3] = v8f{};

  for (int kt = 0; kt < kTiles; ++kt) {
    v16h a = load_a_tile(A, K, arow, kt, lane);
    // speculative prefetch of next A chunk (dropped if invalid per ISA)
    __builtin_prefetch((const void*)(A + (size_t)arow * K + (kt + 1) * 32), 0, 3);
#pragma unroll
    for (int j = 0; j < 4; ++j) {
      v16h b = load_b_tile(W, K, n64 * 64 + j * 16 + (lane & 15), kt, lane);
      acc[j] = __builtin_amdgcn_wmma_f32_16x16x32_f16(
          false, a, false, b, (short)0, acc[j], false, false);
    }
  }

  const int lh = (lane >> 4) & 1;
#pragma unroll
  for (int j = 0; j < 4; ++j) {
    const int n = n64 * 64 + j * 16 + (lane & 15);
    const float bs = bias0[n] + (bias1 ? bias1[n] : 0.0f);
#pragma unroll
    for (int r = 0; r < 8; ++r) {
      const int m = m16 * 16 + r + lh * 8;
      float v = acc[j][r] + bs;
      if (Cf32) {
        Cf32[(size_t)m * N + n] = v;
      } else {
        v = v > 0.0f ? v : 0.0f;
        Cf16[(size_t)m * N + n] = (half_t)v;
      }
    }
  }
}

// ---------------------------------------------------------------------------
// Persistent LSTM layer: 128 blocks x 128 threads run all 512 steps.
//  - w_hh fragments register-resident (128 VGPRs) for the whole sequence
//  - cell state c register-resident (2 floats/thread)
//  - h panel staged to LDS via gfx1250 async global->LDS (guarded fallback)
//  - device-wide step barrier: agent-scope release/acquire atomic + s_sleep
// Block b: m16 = b>>5 (batch tile), ht = b&31 (hidden tile); wave = gate.
// ---------------------------------------------------------------------------
__global__ void __launch_bounds__(128, 1) lstm_layer_persistent(
    const float* __restrict__ gx,       // [512][64][2048] incl. b_ih+b_hh
    const half_t* __restrict__ w_hh,    // [2048][512] f16
    const half_t* __restrict__ h_init,  // [64][512] zeros
    half_t* hs,                         // [512][64][512] all step outputs
    unsigned* sync_cnt) {               // zeroed before launch
  const int tid  = threadIdx.x;
  const int lane = tid & 31;
  const int gate = tid >> 5;                 // 0..3 : i,f,g,o
  const int m16  = blockIdx.x >> 5;          // 0..3
  const int ht   = blockIdx.x & 31;          // 0..31
  const int l15  = lane & 15;
  const int lh   = (lane >> 4) & 1;
  const int nrow = gate * 512 + ht * 16 + l15;
  const int growb = m16 * 16;

  __shared__ half_t sA[16][512];             // 16KB staged h panel
  __shared__ float  gl[4][16][16];           // gate exchange

  // B fragments resident across all steps (16 ktiles x 8 VGPRs = 128 VGPRs)
  v16h barr[16];
#pragma unroll
  for (int kt = 0; kt < 16; ++kt) barr[kt] = load_b_tile(w_hh, 512, nrow, kt, lane);

  // persistent cell state: this thread owns elements tid and 128+tid of the
  // block's 16x16 (row,col) tile.
  float c0 = 0.0f, c1 = 0.0f;
  const int r0 = tid >> 4,          col0 = tid & 15;
  const int r1 = (128 + tid) >> 4,  col1 = tid & 15;

  for (int s = 0; s < 512; ++s) {
    const half_t* hp = (s == 0) ? h_init : (hs + (size_t)(s - 1) * (64 * 512));
    const half_t* base = hp + (size_t)growb * 512;

    // ---- stage 16x512 h panel into LDS (1024 chunks of 16B, 8 per thread)
#ifdef HAVE_ASYNC_LDS
#pragma unroll
    for (int i = 0; i < 8; ++i) {
      const int cidx = i * 128 + tid;        // 0..1023
      const int row = cidx >> 6;
      const int col = (cidx & 63) * 8;
      __builtin_amdgcn_global_load_async_to_lds_b128(
          (as1_i4p)(size_t)(base + row * 512 + col),
          (as3_i4p)(unsigned)(size_t)(&sA[row][col]), 0, 0);
    }
#if __has_builtin(__builtin_amdgcn_s_wait_asynccnt)
    __builtin_amdgcn_s_wait_asynccnt(0);
#else
    asm volatile("s_wait_asynccnt 0x0" ::: "memory");
#endif
#else
#pragma unroll
    for (int i = 0; i < 8; ++i) {
      const int cidx = i * 128 + tid;
      const int row = cidx >> 6;
      const int col = (cidx & 63) * 8;
      *(u32x4*)(&sA[row][col]) = *(const u32x4*)(base + row * 512 + col);
    }
#endif
    __syncthreads();

    // ---- g_tile = h_prev(tile) @ w_hh^T(tile) over K=512, B in registers
    v8f acc = v8f{};
#pragma unroll
    for (int kt = 0; kt < 16; ++kt) {
      const half_t* p = &sA[l15][kt * 32 + lh * 8];
      union { v16h v; u32x4 q[2]; } u;
      u.q[0] = *(const u32x4*)(p);
      u.q[1] = *(const u32x4*)(p + 16);
      acc = __builtin_amdgcn_wmma_f32_16x16x32_f16(
          false, u.v, false, barr[kt], (short)0, acc, false, false);
    }

    // ---- add gx, exchange i,f,g,o through LDS
    const float* gxs = gx + (size_t)s * 64 * 2048;
#pragma unroll
    for (int r = 0; r < 8; ++r) {
      const int row = r + lh * 8;
      gl[gate][row][l15] = acc[r] + gxs[(size_t)(growb + row) * 2048 + nrow];
    }
    __syncthreads();

    // ---- gate math, register-resident c, emit h (f16)
    half_t* hout = hs + (size_t)s * (64 * 512);
    {
      const float gi = gl[0][r0][col0], gf = gl[1][r0][col0];
      const float gg = gl[2][r0][col0], go = gl[3][r0][col0];
      c0 = sigf(gf) * c0 + sigf(gi) * tanhf(gg);
      hout[(size_t)(growb + r0) * 512 + ht * 16 + col0] =
          (half_t)(sigf(go) * tanhf(c0));
    }
    {
      const float gi = gl[0][r1][col1], gf = gl[1][r1][col1];
      const float gg = gl[2][r1][col1], go = gl[3][r1][col1];
      c1 = sigf(gf) * c1 + sigf(gi) * tanhf(gg);
      hout[(size_t)(growb + r1) * 512 + ht * 16 + col1] =
          (half_t)(sigf(go) * tanhf(c1));
    }

    // ---- device-wide step barrier (release h stores, acquire others')
    __threadfence();
    __syncthreads();
    if (tid == 0) {
      const unsigned target = 128u * (unsigned)(s + 1);
      __hip_atomic_fetch_add(sync_cnt, 1u, __ATOMIC_ACQ_REL, __HIP_MEMORY_SCOPE_AGENT);
      while (__hip_atomic_load(sync_cnt, __ATOMIC_ACQUIRE, __HIP_MEMORY_SCOPE_AGENT) < target)
        __builtin_amdgcn_s_sleep(2);
    }
    __syncthreads();
  }
}

// ---------------------------------------------------------------------------
// Final projection: out = hmid @ out_w2^T + b2, sigmoid on cols 24..31.
// M=64, N=32, K=1024 -> 8 waves, 1 block.
// ---------------------------------------------------------------------------
__global__ void __launch_bounds__(256) out_final(
    const half_t* __restrict__ hmid,     // [64][1024] f16 (post relu)
    const half_t* __restrict__ w2,       // [32][1024] f16
    const float* __restrict__ b2, float* __restrict__ out) {  // [64][32]
  const int lane = threadIdx.x & 31;
  const int wave = threadIdx.x >> 5;
  const int m16  = wave >> 1;
  const int n16  = wave & 1;
  const int arow = m16 * 16 + (lane & 15);
  const int nrow = n16 * 16 + (lane & 15);
  v8f acc = v8f{};
#pragma unroll
  for (int kt = 0; kt < 32; ++kt) {
    v16h a = load_a_tile(hmid, 1024, arow, kt, lane);
    v16h b = load_b_tile(w2, 1024, nrow, kt, lane);
    acc = __builtin_amdgcn_wmma_f32_16x16x32_f16(
        false, a, false, b, (short)0, acc, false, false);
  }
  const int lh = (lane >> 4) & 1;
  const int n = n16 * 16 + (lane & 15);
  const float bias = b2[n];
#pragma unroll
  for (int r = 0; r < 8; ++r) {
    const int m = m16 * 16 + r + lh * 8;
    float v = acc[r] + bias;
    if (n >= 24) v = sigf(v);
    out[(size_t)m * 32 + n] = v;
  }
}

// ---------------------------------------------------------------------------
// Build x in [s][b][256] f16: relu(in_w @ ws + in_b) ++ action_emb gather.
// ---------------------------------------------------------------------------
__global__ void __launch_bounds__(256) build_x(
    const float* __restrict__ wst, const int* __restrict__ aidx,
    const float* __restrict__ emb, const float* __restrict__ in_w,
    const float* __restrict__ in_b, half_t* __restrict__ x) {
  const int tid = blockIdx.x * 256 + threadIdx.x;
  const int e = tid & 255;
  const int b = (tid >> 8) & 63;
  const int s = tid >> 14;
  float v;
  if (e < 224) {
    const float* wrow = in_w + e * 32;
    const float* xr = wst + ((size_t)b * 512 + s) * 32;
    float acc = in_b[e];
#pragma unroll
    for (int k = 0; k < 32; ++k) acc += xr[k] * wrow[k];
    v = acc > 0.0f ? acc : 0.0f;
  } else {
    const int ai = aidx[b * 512 + s];
    v = emb[ai * 32 + (e - 224)];
  }
  x[((size_t)s * 64 + b) * 256 + e] = (half_t)v;
}

__global__ void __launch_bounds__(256) cvt_f32_f16(
    const float* __restrict__ s, half_t* __restrict__ d, int n) {
  const int i = blockIdx.x * 256 + threadIdx.x;
  if (i < n) d[i] = (half_t)s[i];
}

__global__ void __launch_bounds__(256) zero_u32(unsigned* __restrict__ p, int n) {
  const int i = blockIdx.x * 256 + threadIdx.x;
  if (i < n) p[i] = 0u;
}

// ---------------------------------------------------------------------------
extern "C" void kernel_launch(void* const* d_in, const int* in_sizes, int n_in,
                              void* d_out, int out_size, void* d_ws, size_t ws_size,
                              hipStream_t stream) {
  (void)in_sizes; (void)n_in; (void)out_size; (void)ws_size;
  const float* wst   = (const float*)d_in[0];
  const int*   aidx  = (const int*)  d_in[1];
  const float* emb   = (const float*)d_in[2];
  const float* in_w  = (const float*)d_in[3];
  const float* in_b  = (const float*)d_in[4];
  const float* w_ih0 = (const float*)d_in[5];
  const float* w_hh0 = (const float*)d_in[6];
  const float* b_ih0 = (const float*)d_in[7];
  const float* b_hh0 = (const float*)d_in[8];
  const float* w_ih1 = (const float*)d_in[9];
  const float* w_hh1 = (const float*)d_in[10];
  const float* b_ih1 = (const float*)d_in[11];
  const float* b_hh1 = (const float*)d_in[12];
  const float* ow1   = (const float*)d_in[13];
  const float* ob1   = (const float*)d_in[14];
  const float* ow2   = (const float*)d_in[15];
  const float* ob2   = (const float*)d_in[16];
  float* out = (float*)d_out;

  char* wp = (char*)d_ws;
  auto carve = [&](size_t bytes) -> char* {
    char* p = wp; wp += (bytes + 255) & ~(size_t)255; return p;
  };
  half_t* w_ih0_h = (half_t*)carve((size_t)2048 * 256 * 2);
  half_t* w_hh0_h = (half_t*)carve((size_t)2048 * 512 * 2);
  half_t* w_ih1_h = (half_t*)carve((size_t)2048 * 512 * 2);
  half_t* w_hh1_h = (half_t*)carve((size_t)2048 * 512 * 2);
  half_t* ow1_h   = (half_t*)carve((size_t)1024 * 512 * 2);
  half_t* ow2_h   = (half_t*)carve((size_t)32 * 1024 * 2);
  half_t* x_h     = (half_t*)carve((size_t)512 * 64 * 256 * 2);   // [s][b][256]
  float*  gx      = (float*) carve((size_t)512 * 64 * 2048 * 4);  // [s*64+b][2048]
  half_t* hs0     = (half_t*)carve((size_t)512 * 64 * 512 * 2);   // [s][b][512]
  half_t* hs1     = (half_t*)carve((size_t)512 * 64 * 512 * 2);   // [s][b][512]
  half_t* h0init  = (half_t*)carve((size_t)64 * 512 * 2);
  half_t* hmid    = (half_t*)carve((size_t)64 * 1024 * 2);
  unsigned* syncb = (unsigned*)carve(256);
  unsigned* sync0 = syncb;
  unsigned* sync1 = syncb + 32;

  // 1) weights -> f16
  cvt_f32_f16<<<2048, 256, 0, stream>>>(w_ih0, w_ih0_h, 2048 * 256);
  cvt_f32_f16<<<4096, 256, 0, stream>>>(w_hh0, w_hh0_h, 2048 * 512);
  cvt_f32_f16<<<4096, 256, 0, stream>>>(w_ih1, w_ih1_h, 2048 * 512);
  cvt_f32_f16<<<4096, 256, 0, stream>>>(w_hh1, w_hh1_h, 2048 * 512);
  cvt_f32_f16<<<2048, 256, 0, stream>>>(ow1, ow1_h, 1024 * 512);
  cvt_f32_f16<<<128, 256, 0, stream>>>(ow2, ow2_h, 32 * 1024);

  // 2) x_h [s][b][256]
  build_x<<<32768, 256, 0, stream>>>(wst, aidx, emb, in_w, in_b, x_h);

  // 3) zeros: initial h, sync counters (re-zeroed every call -> deterministic)
  zero_u32<<<64, 256, 0, stream>>>((unsigned*)h0init, 64 * 512 / 2);
  zero_u32<<<1, 256, 0, stream>>>(syncb, 64);

  // 4) layer-0 gx: [32768 x 2048] = x_h . w_ih0^T + (b_ih0 + b_hh0)
  wmma_gemm_bias<<<8192, 256, 0, stream>>>(x_h, w_ih0_h, b_ih0, b_hh0,
                                           gx, (half_t*)nullptr, 2048, 256, 32);

  // 5) layer-0 recurrence (single persistent kernel, 512 steps)
  lstm_layer_persistent<<<128, 128, 0, stream>>>(gx, w_hh0_h, h0init, hs0, sync0);

  // 6) layer-1 gx (reuse gx): hs0 [32768 x 512] . w_ih1^T + (b_ih1 + b_hh1)
  wmma_gemm_bias<<<8192, 256, 0, stream>>>(hs0, w_ih1_h, b_ih1, b_hh1,
                                           gx, (half_t*)nullptr, 2048, 512, 32);

  // 7) layer-1 recurrence
  lstm_layer_persistent<<<128, 128, 0, stream>>>(gx, w_hh1_h, h0init, hs1, sync1);

  // 8) output MLP on h_T = hs1[511]
  const half_t* hT1 = hs1 + (size_t)511 * 64 * 512;
  wmma_gemm_bias<<<8, 256, 0, stream>>>(hT1, ow1_h, ob1, (const float*)nullptr,
                                        (float*)nullptr, hmid, 1024, 512, 16);
  out_final<<<1, 256, 0, stream>>>(hmid, ow2_h, ob2, out);
}
